// HBNS_73882027425847
// MI455X (gfx1250) — matrix operand order repaired and checked by reading, not compiled, not run on hardware
//
#include <hip/hip_runtime.h>
#include <hip/hip_bf16.h>

typedef __attribute__((ext_vector_type(16))) __bf16          v16bf;
typedef __attribute__((ext_vector_type(16))) unsigned short  v16u16;
typedef __attribute__((ext_vector_type(8)))  unsigned short  u16x8;
typedef __attribute__((ext_vector_type(8)))  float           v8f;

#define FDIM 128

// ---------- helpers ----------

static __device__ __forceinline__ unsigned short f2bf(float f) {
    unsigned u = __float_as_uint(f);
    unsigned r = 0x7FFFu + ((u >> 16) & 1u);   // round-to-nearest-even
    return (unsigned short)((u + r) >> 16);
}
static __device__ __forceinline__ float bf2f(unsigned short h) {
    return __uint_as_float(((unsigned)h) << 16);
}

static __device__ __forceinline__ v8f wmma_bf16(v16bf a, v16bf b, v8f c) {
    return __builtin_amdgcn_wmma_f32_16x16x32_bf16(false, a, false, b, (short)0, c, false, false);
}

// A fragment: 16x32 bf16 from row-major A[m][k] (lda in elements, row stride must be 16B-aligned).
// Lane layout: lanes 0-15 -> M=lane, K = {kb..kb+15}, kb = (lane>>4)*16.
// Each half is 8 contiguous elements -> two b128 loads.
static __device__ __forceinline__ v16bf load_a_frag(const unsigned short* A, int lda,
                                                    int m0, int k0, int lane) {
    int m  = m0 + (lane & 15);
    int kb = ((lane >> 4) & 1) * 16;
    const unsigned short* p = A + (size_t)m * lda + k0 + kb;
    u16x8 lo = *(const u16x8*)(p);
    u16x8 hi = *(const u16x8*)(p + 8);
    v16u16 t = __builtin_shufflevector(lo, hi, 0,1,2,3,4,5,6,7,8,9,10,11,12,13,14,15);
    return __builtin_bit_cast(v16bf, t);
}

// B fragment from FRAGMENT-SWIZZLED weights: element (k,n) stored at
//   ((k>>5)*N + n)*stride + (k&31)
// so each lane's 16 ascending-K values (n = n0+lane&15, K-half h=(lane>>4)) are contiguous.
static __device__ __forceinline__ v16bf load_b_frag(const unsigned short* Wsw, int N, int stride,
                                                    int k0, int n0, int lane) {
    int n = n0 + (lane & 15);
    int h = (lane >> 4) & 1;
    const unsigned short* p = Wsw + ((size_t)(k0 >> 5) * N + n) * stride + h * 16;
    u16x8 lo = *(const u16x8*)(p);
    u16x8 hi = *(const u16x8*)(p + 8);
    v16u16 t = __builtin_shufflevector(lo, hi, 0,1,2,3,4,5,6,7,8,9,10,11,12,13,14,15);
    return __builtin_bit_cast(v16bf, t);
}

static __device__ __forceinline__ void atomicMaxF(float* addr, float val) {
    if (val >= 0.0f) atomicMax((int*)addr, __float_as_int(val));
    else             atomicMin((unsigned int*)addr, __float_as_uint(val));
}
static __device__ __forceinline__ void atomicMinF(float* addr, float val) {
    if (val >= 0.0f) atomicMin((int*)addr, __float_as_int(val));
    else             atomicMax((unsigned int*)addr, __float_as_uint(val));
}

// ---------- kernel: message GEMM  MSG(bf16)[M,128] = X(f32)[M,128] @ W(f32)[128,128] ----------
// 256 threads = 8 waves laid out 4(M)x2(N); block tile 64x128, K=128 in 4 steps of 32.
#define WSTRIDE 40   // per-n element stride in swizzled LDS W (80B: 16B aligned, conflict-free)
__global__ __launch_bounds__(256) void k_gemm_msg(const float* __restrict__ X,
                                                  const float* __restrict__ W,
                                                  unsigned short* __restrict__ MSG, int M) {
    __shared__ unsigned short sWs[4 * FDIM * WSTRIDE];   // swizzled bf16 W
    __shared__ unsigned short sA[64][FDIM + 16];         // bf16 A tile, lda=144 (288B rows)

    const int tid  = threadIdx.x;
    const int lane = tid & 31;
    const int wave = tid >> 5;
    const int wm   = wave & 3;          // M sub-tile (16 rows each)
    const int wn   = wave >> 2;         // N half (64 cols each)

    for (int i = tid; i < FDIM * FDIM; i += 256) {
        int k = i >> 7, n = i & (FDIM - 1);
        sWs[((k >> 5) * FDIM + n) * WSTRIDE + (k & 31)] = f2bf(W[i]);
    }

    for (int tile = blockIdx.x; (size_t)tile * 64 < (size_t)M; tile += gridDim.x) {
        int row0 = tile * 64;
        for (int i = tid; i < 64 * FDIM; i += 256) {
            int r = i >> 7, c = i & (FDIM - 1);
            float v = (row0 + r < M) ? X[(size_t)(row0 + r) * FDIM + c] : 0.0f;
            sA[r][c] = f2bf(v);
        }
        __syncthreads();

        v8f acc[4] = {{0,0,0,0,0,0,0,0},{0,0,0,0,0,0,0,0},{0,0,0,0,0,0,0,0},{0,0,0,0,0,0,0,0}};
#pragma unroll
        for (int k0 = 0; k0 < FDIM; k0 += 32) {
            v16bf a = load_a_frag(&sA[0][0], FDIM + 16, wm * 16, k0, lane);
#pragma unroll
            for (int nt = 0; nt < 4; ++nt) {
                v16bf b = load_b_frag(sWs, FDIM, WSTRIDE, k0, wn * 64 + nt * 16, lane);
                acc[nt] = wmma_bf16(a, b, acc[nt]);
            }
        }

        int mbase = (lane >> 4) * 8;
        int ncol  = lane & 15;
#pragma unroll
        for (int nt = 0; nt < 4; ++nt) {
            int n = wn * 64 + nt * 16 + ncol;
#pragma unroll
            for (int v = 0; v < 8; ++v) {
                int m = row0 + wm * 16 + mbase + v;
                if (m < M) MSG[(size_t)m * FDIM + n] = f2bf(acc[nt][v]);
            }
        }
        __syncthreads();
    }
}

// ---------- kernel: init accumulators ----------
__global__ void k_init(float* __restrict__ s, float* __restrict__ q,
                       float* __restrict__ mx, float* __restrict__ mn,
                       float* __restrict__ cnt, int n) {
    size_t total = (size_t)n * FDIM;
    for (size_t i = (size_t)blockIdx.x * 256 + threadIdx.x; i < total;
         i += (size_t)gridDim.x * 256) {
        s[i]  = 0.0f;
        q[i]  = 0.0f;
        mx[i] = -3.402823466e38f;
        mn[i] =  3.402823466e38f;
        if (i < (size_t)n) cnt[i] = 0.0f;
    }
}

// ---------- kernel: f32 [512][128] -> fragment-swizzled bf16 (L2-resident weights) ----------
__global__ void k_w4_swz(const float* __restrict__ src, unsigned short* __restrict__ dst) {
    for (int i = blockIdx.x * 256 + threadIdx.x; i < 4 * FDIM * FDIM; i += gridDim.x * 256) {
        int k = i >> 7, n = i & (FDIM - 1);
        dst[((size_t)(k >> 5) * FDIM + n) * 32 + (k & 31)] = f2bf(src[i]);
    }
}

// ---------- kernel: edge scatter-aggregate (one wave per edge, 4 feats/lane) ----------
__global__ __launch_bounds__(256) void k_edge_agg(const unsigned short* __restrict__ msg,
                                                  const int* __restrict__ gidx,
                                                  const int* __restrict__ sidx,
                                                  float* __restrict__ a_sum,
                                                  float* __restrict__ a_sq,
                                                  float* __restrict__ a_mx,
                                                  float* __restrict__ a_mn,
                                                  float* __restrict__ a_cnt, int nnz) {
    int e = blockIdx.x * 8 + (threadIdx.x >> 5);
    if (e >= nnz) return;
    int lane = threadIdx.x & 31;
    int g = gidx[e];
    int s = sidx[e];
    if (lane == 0) atomicAdd(&a_cnt[s], 1.0f);

    const uint2* mrow = (const uint2*)(msg + (size_t)g * FDIM);   // 8B/lane: global_load_b64
    uint2 d = mrow[lane];
    float v0 = bf2f((unsigned short)(d.x & 0xffffu));
    float v1 = bf2f((unsigned short)(d.x >> 16));
    float v2 = bf2f((unsigned short)(d.y & 0xffffu));
    float v3 = bf2f((unsigned short)(d.y >> 16));

    size_t base = (size_t)s * FDIM + lane * 4;
    float vv[4] = {v0, v1, v2, v3};
#pragma unroll
    for (int j = 0; j < 4; ++j) {
        float v = vv[j];
        atomicAdd(&a_sum[base + j], v);
        atomicAdd(&a_sq[base + j], v * v);
        atomicMaxF(&a_mx[base + j], v);
        atomicMinF(&a_mn[base + j], v);
    }
}

// ---------- kernel: fused PNA stats -> [16,512]@[512,128] WMMA -> bias+relu -> LN -> relu ----------
// 256 threads = 8 waves; each wave owns one 16x16 output tile (n-col = wave*16).
__global__ __launch_bounds__(256) void k_pna_ln(const float* __restrict__ a_sum,
                                                const float* __restrict__ a_sq,
                                                const float* __restrict__ a_mx,
                                                const float* __restrict__ a_mn,
                                                const float* __restrict__ a_cnt,
                                                const unsigned short* __restrict__ W4sw, // swizzled bf16
                                                const float* __restrict__ bias,
                                                const float* __restrict__ gamma,
                                                const float* __restrict__ beta,
                                                float* __restrict__ out, int n) {
    __shared__ unsigned short sFeat[16][512 + 16];  // bf16 [mean|min|max|std], 1056B rows
    __shared__ float sY[16][FDIM];                  // post-GEMM activations

    const int tid  = threadIdx.x;
    const int lane = tid & 31;
    const int wave = tid >> 5;

    for (int t = blockIdx.x; (size_t)t * 16 < (size_t)n; t += gridDim.x) {
        int node0 = t * 16;

        // build PNA stats tile
        for (int i = tid; i < 16 * FDIM; i += 256) {
            int r = i >> 7, f = i & (FDIM - 1);
            int node = node0 + r;
            float mean = 0.0f, vmn = 0.0f, vmx = 0.0f, stdv = 0.0f;
            if (node < n) {
                float c = a_cnt[node];
                float denom = fmaxf(c, 1.0f);
                size_t b = (size_t)node * FDIM + f;
                float s = a_sum[b], q = a_sq[b];
                mean = s / denom;
                float var = fmaxf(q / denom - mean * mean, 0.0f);
                stdv = sqrtf(var + 1e-5f);
                bool has = c > 0.0f;
                vmx = has ? a_mx[b] : 0.0f;
                vmn = has ? a_mn[b] : 0.0f;
            }
            sFeat[r][f]          = f2bf(mean);
            sFeat[r][FDIM + f]   = f2bf(vmn);
            sFeat[r][2*FDIM + f] = f2bf(vmx);
            sFeat[r][3*FDIM + f] = f2bf(stdv);
        }
        __syncthreads();

        // K=512 GEMM; B fragments are contiguous b128 loads from L2-resident swizzled W4
        v8f acc = {0,0,0,0,0,0,0,0};
#pragma unroll
        for (int k0 = 0; k0 < 512; k0 += 32) {
            v16bf a = load_a_frag(&sFeat[0][0], 512 + 16, 0, k0, lane);
            v16bf b = load_b_frag(W4sw, FDIM, 32, k0, wave * 16, lane);
            acc = wmma_bf16(a, b, acc);
        }

        int mbase = (lane >> 4) * 8;
        int ncol  = wave * 16 + (lane & 15);
        float bb  = bias[ncol];
#pragma unroll
        for (int v = 0; v < 8; ++v)
            sY[mbase + v][ncol] = fmaxf(acc[v] + bb, 0.0f);   // pna relu
        __syncthreads();

        // layernorm + relu: wave w handles rows w and w+8, 4 feats/lane, shuffle reduction
#pragma unroll
        for (int rep = 0; rep < 2; ++rep) {
            int r = wave + rep * 8;
            int node = node0 + r;
            float vals[4];
            float s1 = 0.0f, s2 = 0.0f;
#pragma unroll
            for (int j = 0; j < 4; ++j) {
                float v = sY[r][lane + 32 * j];
                vals[j] = v; s1 += v; s2 += v * v;
            }
#pragma unroll
            for (int off = 16; off > 0; off >>= 1) {
                s1 += __shfl_xor(s1, off, 32);
                s2 += __shfl_xor(s2, off, 32);
            }
            float mu   = s1 * (1.0f / 128.0f);
            float var  = fmaxf(s2 * (1.0f / 128.0f) - mu * mu, 0.0f);
            float rstd = rsqrtf(var + 1e-5f);
            if (node < n) {
#pragma unroll
                for (int j = 0; j < 4; ++j) {
                    int f = lane + 32 * j;
                    float o = (vals[j] - mu) * rstd * gamma[f] + beta[f];
                    out[(size_t)node * FDIM + f] = fmaxf(o, 0.0f);
                }
            }
        }
        __syncthreads();
    }
}

// ---------- host ----------
extern "C" void kernel_launch(void* const* d_in, const int* in_sizes, int n_in,
                              void* d_out, int out_size, void* d_ws, size_t ws_size,
                              hipStream_t stream) {
    const float* x_source  = (const float*)d_in[0];
    const float* x_target  = (const float*)d_in[1];
    const float* w_s       = (const float*)d_in[2];
    const float* w_t       = (const float*)d_in[3];
    // d_in[4], d_in[5]: dead cci branches (results discarded in reference)
    const float* pna_src_w = (const float*)d_in[6];
    const float* pna_src_b = (const float*)d_in[7];
    const float* pna_tgt_w = (const float*)d_in[8];
    const float* pna_tgt_b = (const float*)d_in[9];
    const float* ln_g      = (const float*)d_in[10];
    const float* ln_b      = (const float*)d_in[11];
    const int*   rows      = (const int*)d_in[12];
    const int*   cols      = (const int*)d_in[13];

    const int NS  = in_sizes[0] / FDIM;
    const int NT  = in_sizes[1] / FDIM;
    const int NNZ = in_sizes[12];
    const int NMX = (NS > NT) ? NS : NT;

    // workspace carve-up (accumulators sized for max(NS,NT), reused across phases)
    char* ws = (char*)d_ws;
    size_t off = 0;
    auto take = [&](size_t bytes) -> char* {
        char* p = ws + off;
        off += (bytes + 255) & ~(size_t)255;
        return p;
    };
    unsigned short* msg   = (unsigned short*)take((size_t)NMX * FDIM * sizeof(unsigned short));
    float*          a_sum = (float*)take((size_t)NMX * FDIM * sizeof(float));
    float*          a_sq  = (float*)take((size_t)NMX * FDIM * sizeof(float));
    float*          a_mx  = (float*)take((size_t)NMX * FDIM * sizeof(float));
    float*          a_mn  = (float*)take((size_t)NMX * FDIM * sizeof(float));
    float*          a_cnt = (float*)take((size_t)NMX * sizeof(float));
    unsigned short* w4sw  = (unsigned short*)take((size_t)4 * FDIM * FDIM * sizeof(unsigned short));
    (void)ws_size; (void)n_in; (void)out_size;

    float* out_src = (float*)d_out;
    float* out_tgt = out_src + (size_t)NS * FDIM;

    const int W4N = 4 * FDIM * FDIM;

    // ---- phase 1: out_src  (t_message gathered by rows, segmented by cols -> NS) ----
    k_gemm_msg<<<(NT + 63) / 64, 256, 0, stream>>>(x_target, w_t, msg, NT);
    k_init<<<((size_t)NS * FDIM + 255) / 256, 256, 0, stream>>>(a_sum, a_sq, a_mx, a_mn, a_cnt, NS);
    k_w4_swz<<<(W4N + 255) / 256, 256, 0, stream>>>(pna_src_w, w4sw);
    k_edge_agg<<<(NNZ + 7) / 8, 256, 0, stream>>>(msg, rows, cols, a_sum, a_sq, a_mx, a_mn, a_cnt, NNZ);
    k_pna_ln<<<(NS + 15) / 16, 256, 0, stream>>>(a_sum, a_sq, a_mx, a_mn, a_cnt,
                                                 w4sw, pna_src_b, ln_g, ln_b, out_src, NS);

    // ---- phase 2: out_tgt  (s_message gathered by cols, segmented by rows -> NT) ----
    k_gemm_msg<<<(NS + 63) / 64, 256, 0, stream>>>(x_source, w_s, msg, NS);
    k_init<<<((size_t)NT * FDIM + 255) / 256, 256, 0, stream>>>(a_sum, a_sq, a_mx, a_mn, a_cnt, NT);
    k_w4_swz<<<(W4N + 255) / 256, 256, 0, stream>>>(pna_tgt_w, w4sw);
    k_edge_agg<<<(NNZ + 7) / 8, 256, 0, stream>>>(msg, cols, rows, a_sum, a_sq, a_mx, a_mn, a_cnt, NNZ);
    k_pna_ln<<<(NT + 15) / 16, 256, 0, stream>>>(a_sum, a_sq, a_mx, a_mn, a_cnt,
                                                 w4sw, pna_tgt_b, ln_g, ln_b, out_tgt, NT);
}
